// ScaledDotProductAttention_25838523253001
// MI455X (gfx1250) — compile-verified, matrix operand order checked by reading, not run on hardware
//
#include <hip/hip_runtime.h>

// ---------------------------------------------------------------------------
// Causal MHA (QKV proj -> flash attention -> output mixer) for MI455X gfx1250.
// All matmuls on v_wmma_f32_16x16x32_bf16 with a hi/lo bf16 split of fp32 data
// (3 WMMAs per 32-k step: hi*hi + hi*lo + lo*hi => ~fp32 accuracy).
// LDS tiles are stored as separate hi/lo bf16 planes, PRE-PERMUTED into the
// WMMA operand element order, so each operand is a single contiguous 32B LDS
// read bit-cast to v16bf -- no unpack VALU in the inner loops.
// Workspace (~134.3 MB): QKV [8192][3072] f32, attnO [8192][1024] f32.
// ---------------------------------------------------------------------------

typedef __attribute__((ext_vector_type(16))) __bf16          v16bf;
typedef __attribute__((ext_vector_type(16))) unsigned short  v16us;
typedef __attribute__((ext_vector_type(8)))  float           v8f;
typedef __attribute__((ext_vector_type(8)))  unsigned        v8u;

#define B_  4
#define W_  2048
#define C_  1024
#define H_  16
#define KD  64

// round-to-nearest-even fp32 -> bf16 (top 16 bits)
__device__ __forceinline__ unsigned bf16_rne(float x) {
  unsigned u = __builtin_bit_cast(unsigned, x);
  return (u + 0x7FFFu + ((u >> 16) & 1u)) >> 16;
}
__device__ __forceinline__ float bf16_hi_f(unsigned h) {
  return __builtin_bit_cast(float, h << 16);
}
// split two adjacent floats into packed-pair hi plane / lo plane dwords
__device__ __forceinline__ void split2(float x0, float x1, unsigned& hp, unsigned& lp) {
  unsigned h0 = bf16_rne(x0), h1 = bf16_rne(x1);
  unsigned l0 = bf16_rne(x0 - bf16_hi_f(h0));
  unsigned l1 = bf16_rne(x1 - bf16_hi_f(h1));
  hp = h0 | (h1 << 16);
  lp = l0 | (l1 << 16);
}
__device__ __forceinline__ void split1(float x, unsigned short& h, unsigned short& l) {
  unsigned hh = bf16_rne(x);
  h = (unsigned short)hh;
  l = (unsigned short)bf16_rne(x - bf16_hi_f(hh));
}

// WMMA element order within one 32-k window (bf16 position for k):
//   pos = ((k>>3)&1)*16 + (k>>4)*8 + (k&7)      (lane half g reads pos g*16..)
// dword index for an even-k pair:
__device__ __forceinline__ int perm_dw(int kk) {
  return ((kk >> 3) & 1) * 8 + ((kk >> 4) << 2) + ((kk & 7) >> 1);
}
__device__ __forceinline__ int perm_b16(int kk) {
  return ((kk >> 3) & 1) * 16 + ((kk >> 4) << 3) + (kk & 7);
}

// One operand = 16 bf16 = 32 contiguous bytes in LDS (two ds_load_b128).
__device__ __forceinline__ v16bf load_opnd(const unsigned* p) {
  v8u raw = *(const v8u*)p;
  return __builtin_bit_cast(v16bf, raw);
}

// C += Ah*Bh + Ah*Bl + Al*Bh
__device__ __forceinline__ v8f wmma_bf16x3(v16bf ah, v16bf al, v16bf bh, v16bf bl, v8f c) {
  c = __builtin_amdgcn_wmma_f32_16x16x32_bf16(false, ah, false, bh, (short)0, c, false, false);
  c = __builtin_amdgcn_wmma_f32_16x16x32_bf16(false, ah, false, bl, (short)0, c, false, false);
  c = __builtin_amdgcn_wmma_f32_16x16x32_bf16(false, al, false, bh, (short)0, c, false, false);
  return c;
}

// ---------------------------------------------------------------------------
// GEMM: C[M][N] = A[M][K] * B (+bias). bT==0: B is [K][N]. bT==1: B is [N][K]
// (A @ B^T). 256 threads (8 waves), 128x128 tile, k-step 32.
// ---------------------------------------------------------------------------
#define GRS 24  // row stride in dwords (16 used + 8 pad; multiple of 8 => 32B align)

__global__ __launch_bounds__(256) void gemm_bf16x3_kernel(
    const float* __restrict__ A, const float* __restrict__ Bm,
    const float* __restrict__ bias, float* __restrict__ Cm,
    int M, int N, int Kdim, int bT) {
  __shared__ __align__(32) unsigned sAhi[128 * GRS];
  __shared__ __align__(32) unsigned sAlo[128 * GRS];
  __shared__ __align__(32) unsigned sBhi[128 * GRS];
  __shared__ __align__(32) unsigned sBlo[128 * GRS];

  const int tid  = threadIdx.x;
  const int m0   = blockIdx.y * 128;
  const int n0   = blockIdx.x * 128;
  const int wave = tid >> 5;
  const int lane = tid & 31;
  const int g    = lane >> 4;
  const int r    = lane & 15;

  v8f acc[8];
#pragma unroll
  for (int t = 0; t < 8; ++t) acc[t] = {};

  const int ar  = tid >> 1;         // 0..127 (A rows / bT==1 B rows)
  const int ak  = (tid & 1) * 16;   // 0 or 16
  const int bkr = tid >> 3;         // 0..31  (bT==0 B k-rows)
  const int bn0 = (tid & 7) * 16;

  for (int k0 = 0; k0 < Kdim; k0 += 32) {
    __syncthreads();
    {  // stage A rows: contiguous k -> permuted pair stores (ds_store_b32)
      const float* ap = A + (size_t)(m0 + ar) * Kdim + k0 + ak;
      unsigned* dh = sAhi + ar * GRS;
      unsigned* dl = sAlo + ar * GRS;
#pragma unroll
      for (int i = 0; i < 8; ++i) {
        const int kk = ak + 2 * i;
        unsigned hp, lp;
        split2(ap[2 * i], ap[2 * i + 1], hp, lp);
        const int d = perm_dw(kk);
        dh[d] = hp;
        dl[d] = lp;
      }
      if (k0 + 32 < Kdim) __builtin_prefetch(ap + 32, 0, 1);  // global_prefetch_b8
    }
    if (bT == 0) {  // B[k][n]: transpose-stage, b16 scatters
      const float* bp = Bm + (size_t)(k0 + bkr) * N + n0 + bn0;
      unsigned short* dh = (unsigned short*)sBhi;
      unsigned short* dl = (unsigned short*)sBlo;
      const int pos = perm_b16(bkr);
#pragma unroll
      for (int i = 0; i < 16; ++i) {
        unsigned short h, l;
        split1(bp[i], h, l);
        dh[(bn0 + i) * (2 * GRS) + pos] = h;
        dl[(bn0 + i) * (2 * GRS) + pos] = l;
      }
    } else {        // B[n][k]: contiguous k, pair stores
      const float* bp = Bm + (size_t)(n0 + ar) * Kdim + k0 + ak;
      unsigned* dh = sBhi + ar * GRS;
      unsigned* dl = sBlo + ar * GRS;
#pragma unroll
      for (int i = 0; i < 8; ++i) {
        const int kk = ak + 2 * i;
        unsigned hp, lp;
        split2(bp[2 * i], bp[2 * i + 1], hp, lp);
        const int d = perm_dw(kk);
        dh[d] = hp;
        dl[d] = lp;
      }
    }
    __syncthreads();

    const int arow = (wave * 16 + r) * GRS + g * 8;
    const v16bf ah = load_opnd(sAhi + arow);
    const v16bf al = load_opnd(sAlo + arow);
#pragma unroll
    for (int t = 0; t < 8; ++t) {
      const int brow = (t * 16 + r) * GRS + g * 8;
      const v16bf bh = load_opnd(sBhi + brow);
      const v16bf bl = load_opnd(sBlo + brow);
      acc[t] = wmma_bf16x3(ah, al, bh, bl, acc[t]);
    }
  }

  // Epilogue: C layout -> row = v + 8*g, col = r
#pragma unroll
  for (int t = 0; t < 8; ++t) {
    const int gn = n0 + t * 16 + r;
    const float bv = bias ? bias[gn] : 0.0f;
#pragma unroll
    for (int v = 0; v < 8; ++v) {
      const int gm = m0 + wave * 16 + v + 8 * g;
      Cm[(size_t)gm * N + gn] = acc[t][v] + bv;
    }
  }
}

// ---------------------------------------------------------------------------
// Causal flash attention. Block = (b*H+h, 64-row q block), 128 threads =
// 4 waves; each wave owns 16 query rows. s-tiles of 32.
// ---------------------------------------------------------------------------
#define KRS 40  // sK row stride, dwords (2 windows * 16 + 8 pad; mult of 8)
#define VRS 24  // sV row stride, dwords
#define PRS 24  // sP row stride, dwords

__global__ __launch_bounds__(128) void flash_attn_kernel(
    const float* __restrict__ qkv, float* __restrict__ Om) {
  __shared__ __align__(32) unsigned sKhi[32 * KRS];
  __shared__ __align__(32) unsigned sKlo[32 * KRS];
  __shared__ __align__(32) unsigned sVhi[64 * VRS];
  __shared__ __align__(32) unsigned sVlo[64 * VRS];
  __shared__ __align__(32) unsigned sPhi[4 * 16 * PRS];
  __shared__ __align__(32) unsigned sPlo[4 * 16 * PRS];

  const int tid  = threadIdx.x;
  const int wave = tid >> 5;
  const int lane = tid & 31;
  const int g    = lane >> 4;
  const int r    = lane & 15;
  const int b    = blockIdx.x / H_;
  const int h    = blockIdx.x % H_;
  const int q0   = blockIdx.y * 64;
  const int qw   = q0 + wave * 16;

  // ---- Q into registers, WMMA A-layout, split hi/lo, two k-steps of 32 ----
  v16bf qh[2], ql[2];
  {
    const float* gq = qkv + (size_t)(b * W_ + qw + r) * (3 * C_) + h * KD;
#pragma unroll
    for (int t = 0; t < 2; ++t) {
      v16us hu, lu;
#pragma unroll
      for (int e = 0; e < 16; ++e) {
        const int k = t * 32 + (e >> 3) * 16 + g * 8 + (e & 7);
        unsigned short hh, ll;
        split1(gq[k], hh, ll);
        hu[e] = hh;
        lu[e] = ll;
      }
      qh[t] = __builtin_bit_cast(v16bf, hu);
      ql[t] = __builtin_bit_cast(v16bf, lu);
    }
  }

  float mrow[8], lrow[8];
#pragma unroll
  for (int v = 0; v < 8; ++v) { mrow[v] = -1e30f; lrow[v] = 0.0f; }
  v8f o[4];
#pragma unroll
  for (int t = 0; t < 4; ++t) o[t] = {};

  const int sr   = tid >> 2;        // staged s-row 0..31
  const int sk   = (tid & 3) * 16;  // staged k-offset 0/16/32/48
  const int send = q0 + 64;

  for (int s0 = 0; s0 < send; s0 += 32) {
    __syncthreads();
    {  // stage K (permuted pairs) and V (transposed b16 scatter)
      const size_t rb = (size_t)(b * W_ + s0 + sr) * (3 * C_) + h * KD + sk;
      const float* kp = qkv + rb + C_;
      const float* vp = qkv + rb + 2 * C_;
      unsigned* dh = sKhi + sr * KRS;
      unsigned* dl = sKlo + sr * KRS;
#pragma unroll
      for (int i = 0; i < 8; ++i) {
        const int kk = sk + 2 * i;           // 0..63
        unsigned hp, lp;
        split2(kp[2 * i], kp[2 * i + 1], hp, lp);
        const int d = (kk >> 5) * 16 + perm_dw(kk & 31);
        dh[d] = hp;
        dl[d] = lp;
      }
      unsigned short* vh = (unsigned short*)sVhi;
      unsigned short* vl = (unsigned short*)sVlo;
      const int pos = perm_b16(sr);          // s position within window
#pragma unroll
      for (int i = 0; i < 16; ++i) {
        unsigned short hh, ll;
        split1(vp[i], hh, ll);
        vh[(sk + i) * (2 * VRS) + pos] = hh;
        vl[(sk + i) * (2 * VRS) + pos] = ll;
      }
    }
    __syncthreads();

    if (s0 <= qw + 15) {  // wave-uniform branch: EXEC all-ones for WMMA
      // ---- S = Q K^T : two 16-col tiles, k = 64 = 2 windows ----
      v8f s[2];
#pragma unroll
      for (int j = 0; j < 2; ++j) {
        v8f a = {};
#pragma unroll
        for (int t = 0; t < 2; ++t) {
          const int krow = (j * 16 + r) * KRS + t * 16 + g * 8;
          a = wmma_bf16x3(qh[t], ql[t], load_opnd(sKhi + krow), load_opnd(sKlo + krow), a);
        }
        s[j] = a;
      }
      // ---- scale, causal mask, online softmax ----
      float alpha[8];
#pragma unroll
      for (int v = 0; v < 8; ++v) {
        const int qg = qw + v + 8 * g;
        float x0 = (s0 + r      <= qg) ? s[0][v] * 0.125f : -1e30f;
        float x1 = (s0 + 16 + r <= qg) ? s[1][v] * 0.125f : -1e30f;
        float tm = fmaxf(x0, x1);
#pragma unroll
        for (int msk = 1; msk < 16; msk <<= 1) tm = fmaxf(tm, __shfl_xor(tm, msk, 32));
        const float nm = fmaxf(mrow[v], tm);
        alpha[v] = __expf(mrow[v] - nm);
        mrow[v]  = nm;
        const float p0 = __expf(x0 - nm);
        const float p1 = __expf(x1 - nm);
        s[0][v] = p0;
        s[1][v] = p1;
        float rs = p0 + p1;
#pragma unroll
        for (int msk = 1; msk < 16; msk <<= 1) rs += __shfl_xor(rs, msk, 32);
        lrow[v] = lrow[v] * alpha[v] + rs;
      }
#pragma unroll
      for (int t = 0; t < 4; ++t)
#pragma unroll
        for (int v = 0; v < 8; ++v) o[t][v] *= alpha[v];

      // ---- P: C-layout -> per-wave LDS planes (b16 scatter, permuted) ----
      unsigned short* ph16 = (unsigned short*)(sPhi + wave * 16 * PRS);
      unsigned short* pl16 = (unsigned short*)(sPlo + wave * 16 * PRS);
#pragma unroll
      for (int v = 0; v < 8; ++v) {
        const int row = (v + 8 * g) * (2 * PRS);
#pragma unroll
        for (int j = 0; j < 2; ++j) {
          const int pos = (r >> 3) * 16 + j * 8 + (r & 7);  // perm of s=j*16+r
          unsigned short hh, ll;
          split1(s[j][v], hh, ll);
          ph16[row + pos] = hh;
          pl16[row + pos] = ll;
        }
      }
      const int prow = wave * 16 * PRS + r * PRS + g * 8;
      const v16bf pH = load_opnd(sPhi + prow);
      const v16bf pL = load_opnd(sPlo + prow);

      // ---- O += P V : 4 output tiles over head-dim 64 ----
#pragma unroll
      for (int t = 0; t < 4; ++t) {
        const int vrow = (t * 16 + r) * VRS + g * 8;
        o[t] = wmma_bf16x3(pH, pL, load_opnd(sVhi + vrow), load_opnd(sVlo + vrow), o[t]);
      }
    }
  }

  // ---- normalize and store O[b][q][h*64 + kcol] ----
#pragma unroll
  for (int v = 0; v < 8; ++v) {
    const float inv = 1.0f / lrow[v];
    const int q = qw + v + 8 * g;
    float* op = Om + (size_t)(b * W_ + q) * C_ + h * KD + r;
#pragma unroll
    for (int t = 0; t < 4; ++t) op[t * 16] = o[t][v] * inv;
  }
}

// ---------------------------------------------------------------------------
extern "C" void kernel_launch(void* const* d_in, const int* in_sizes, int n_in,
                              void* d_out, int out_size, void* d_ws, size_t ws_size,
                              hipStream_t stream) {
  (void)in_sizes; (void)n_in; (void)out_size; (void)ws_size;
  const float* x    = (const float*)d_in[0];   // [4,2048,1024]
  const float* wqkv = (const float*)d_in[1];   // [1024,3072]
  const float* wmix = (const float*)d_in[2];   // [1024,1024]
  const float* bmix = (const float*)d_in[3];   // [1024]
  float* out = (float*)d_out;

  float* qkv = (float*)d_ws;                        // [8192][3072]
  float* ao  = qkv + (size_t)(B_ * W_) * (3 * C_);  // [8192][1024]
  const int M = B_ * W_;

  gemm_bf16x3_kernel<<<dim3((3 * C_) / 128, M / 128), 256, 0, stream>>>(
      x, wqkv, nullptr, qkv, M, 3 * C_, C_, 0);
  flash_attn_kernel<<<dim3(B_ * H_, W_ / 64), 128, 0, stream>>>(qkv, ao);
  gemm_bf16x3_kernel<<<dim3(C_ / 128, M / 128), 256, 0, stream>>>(
      ao, wmix, bmix, out, M, C_, C_, 1);
}
